// WindowAttention_11510512353811
// MI455X (gfx1250) — matrix operand order repaired and checked by reading, not compiled, not run on hardware
//
#include <hip/hip_runtime.h>
#include <hip/hip_bf16.h>

typedef __attribute__((ext_vector_type(16))) _Float16 v16h;
typedef __attribute__((ext_vector_type(8)))  _Float16 v8h;
typedef __attribute__((ext_vector_type(8)))  float    v8f;

#define TOK  64
#define CH   256
#define HD   32
#define NH   8
#define NWIN 1024

__device__ __forceinline__ int lane_id() { return (int)(threadIdx.x & 31u); }

__device__ __forceinline__ v8f wmma16(v16h a, v16h b, v8f c) {
  // D = A(16x32 f16) * B(32x16 f16) + C(16x16 f32)
  return __builtin_amdgcn_wmma_f32_16x16x32_f16(false, a, false, b, (short)0, c,
                                                false, false);
}

// ---- WMMA fragment helpers (layouts per CDNA5 ISA 7.12.2, wave32) ----

// A fragment 16x32 f16 from row-major f16 matrix; base -> (row0, k0), leading dim ld.
// lanes 0-15: row M=lane, K = 0..7 & 16..23 ; lanes 16-31: row M=lane-16, K = 8..15 & 24..31
__device__ __forceinline__ v16h load_a_f16(const _Float16* base, int ld) {
  const int lane = lane_id();
  const _Float16* p = base + (lane & 15) * ld + ((lane >> 4) << 3);
  v8h lo = *(const v8h*)p;        // K = kb .. kb+7
  v8h hi = *(const v8h*)(p + 16); // K = kb+16 .. kb+23
  v16h r;
#pragma unroll
  for (int i = 0; i < 8; ++i) { r[i] = lo[i]; r[i + 8] = hi[i]; }
  return r;
}

// A fragment from row-major f32 source (attention probabilities), converting to f16.
__device__ __forceinline__ v16h load_a_f32(const float* base, int ld) {
  const int lane = lane_id();
  const float* p = base + (lane & 15) * ld + ((lane >> 4) << 3);
  v16h r;
#pragma unroll
  for (int i = 0; i < 8; ++i) {
    r[i]     = (_Float16)p[i];
    r[i + 8] = (_Float16)p[i + 16];
  }
  return r;
}

// B fragment 32x16 f16 where B[k][n] = W[n][k] (W row-major, leading dim ld).
// lanes 0-15: col N=lane, K = 0..7 & 16..23 ; lanes 16-31: col N=lane-16, K = 8..15 & 24..31
__device__ __forceinline__ v16h load_b_t(const _Float16* base, int ld) {
  const int lane = lane_id();
  const _Float16* p = base + (lane & 15) * ld + ((lane >> 4) << 3);
  v8h lo = *(const v8h*)p;
  v8h hi = *(const v8h*)(p + 16);
  v16h r;
#pragma unroll
  for (int i = 0; i < 8; ++i) { r[i] = lo[i]; r[i + 8] = hi[i]; }
  return r;
}

// Store D (16x16 f32 acc) to f16 row-major [ld], scaled.
// VGPR r holds (M = r + 8*(lane>=16), N = lane&15)
__device__ __forceinline__ void store_d_f16(_Float16* base, int ld, v8f d, float s) {
  const int lane = lane_id();
  const int col = lane & 15;
  const int rb  = (lane >> 4) << 3;
#pragma unroll
  for (int r = 0; r < 8; ++r) base[(rb + r) * ld + col] = (_Float16)(d[r] * s);
}

// Transposed store: element (row,col) -> base[col*ld + row]   (used for V^T)
__device__ __forceinline__ void store_d_f16_t(_Float16* base, int ld, v8f d) {
  const int lane = lane_id();
  const int col = lane & 15;
  const int rb  = (lane >> 4) << 3;
#pragma unroll
  for (int r = 0; r < 8; ++r) base[col * ld + rb + r] = (_Float16)d[r];
}

// ---- Prep: fp32 weights -> f16, expand relative-position bias table ----
__global__ void winattn_prep(const float* __restrict__ qkv_w,
                             const float* __restrict__ proj_w,
                             const float* __restrict__ bias_table,
                             _Float16* __restrict__ qkvw_h,
                             _Float16* __restrict__ projw_h,
                             float* __restrict__ relBias) {
  const int NQ = 3 * CH * CH;   // 196608
  const int NP = CH * CH;       // 65536
  const int NR = NH * TOK * TOK;// 32768
  int i = blockIdx.x * blockDim.x + threadIdx.x;
  if (i < NQ) {
    qkvw_h[i] = (_Float16)qkv_w[i];
  } else if (i < NQ + NP) {
    int j = i - NQ;
    projw_h[j] = (_Float16)proj_w[j];
  } else if (i < NQ + NP + NR) {
    int j = i - NQ - NP;
    int h = j >> 12;            // / 4096
    int n = (j >> 6) & 63;
    int m = j & 63;
    relBias[j] = bias_table[(n - m + (TOK - 1)) * NH + h];
  }
}

// ---- Main: one block per window, 8 waves, wave w owns head w ----
__global__ __launch_bounds__(256, 1)
void winattn_main(const float* __restrict__ x,
                  const float* __restrict__ mask,
                  const _Float16* __restrict__ qkvw,
                  const float* __restrict__ qkvb,
                  const _Float16* __restrict__ projw,
                  const float* __restrict__ projb,
                  const float* __restrict__ relBias,
                  float* __restrict__ out) {
  extern __shared__ char smem[];
  _Float16* Xh = (_Float16*)smem;        // [64][256] f16, later reused as attn out O
  _Float16* Qh = Xh + TOK * CH;          // [64][256] f16 (pre-scaled)
  _Float16* Kh = Qh + TOK * CH;          // [64][256] f16
  _Float16* Vt = Kh + TOK * CH;          // [256][64] f16 (transposed V)
  float*  aBuf = (float*)(Vt + TOK * CH);// [8][64][64] f32

  const int b    = blockIdx.x;
  const int tid  = (int)threadIdx.x;
  const int wave = tid >> 5;
  const int lane = tid & 31;
  const int col  = lane & 15;
  const int rbase = (lane >> 4) << 3;
  const float* xb = x + (size_t)b * TOK * CH;

  if (tid == 0) {
    __builtin_prefetch(qkvw, 0, 3);   // global_prefetch: pull weights toward L2
    __builtin_prefetch(projw, 0, 3);
  }

  // Stage A: x[b] (fp32) -> f16 LDS
  const float4* x4 = (const float4*)xb;
  for (int i = tid; i < (TOK * CH) / 4; i += 256) {
    float4 v = x4[i];
    int o = i * 4;
    Xh[o + 0] = (_Float16)v.x;
    Xh[o + 1] = (_Float16)v.y;
    Xh[o + 2] = (_Float16)v.z;
    Xh[o + 3] = (_Float16)v.w;
  }
  __syncthreads();

  // Stage B: QKV GEMM  [64 x 768] = Xh[64x256] * qkv_w^T + bias.
  // mt outer: A-fragments for row-block mt stay resident in VGPRs and are
  // reused across all 6 N-tiles owned by this wave. nt loop kept ROLLED
  // (unroll 1) so only one iteration's B-fragments are live -> no spills.
  const float scale = 0.17677669529663687f; // 32^-0.5
  float biasq[6];
#pragma unroll
  for (int i = 0; i < 6; ++i) biasq[i] = qkvb[(wave + NH * i) * 16 + col];

#pragma unroll 1
  for (int mt = 0; mt < 4; ++mt) {
    v16h aF[8];
#pragma unroll
    for (int kk = 0; kk < 8; ++kk)
      aF[kk] = load_a_f16(Xh + mt * 16 * CH + kk * 32, CH);
#pragma unroll 1
    for (int i = 0; i < 6; ++i) {
      const int nt = wave + NH * i;
      v8f acc;
#pragma unroll
      for (int r = 0; r < 8; ++r) acc[r] = biasq[i];
#pragma unroll
      for (int kk = 0; kk < 8; ++kk) {
        v16h bw = load_b_t(qkvw + (size_t)(nt * 16) * CH + kk * 32, CH);
        acc = wmma16(aF[kk], bw, acc);
      }
      if (nt < 16) {
        store_d_f16(Qh + mt * 16 * CH + nt * 16, CH, acc, scale);
      } else if (nt < 32) {
        store_d_f16(Kh + mt * 16 * CH + (nt - 16) * 16, CH, acc, 1.0f);
      } else {
        // V^T[channel][token]
        store_d_f16_t(Vt + ((nt - 32) * 16) * TOK + mt * 16, TOK, acc);
      }
    }
  }
  __syncthreads();

  // Stage C: attn = Q_h * K_h^T + rel_bias + mask   (wave w == head w)
  {
    const int h = wave;
    float* aB = aBuf + h * TOK * TOK;
    const float* rb_h = relBias + h * TOK * TOK;
    const float* mk   = mask + (size_t)(b & (NWIN - 1)) * TOK * TOK;
#pragma unroll 1
    for (int mt = 0; mt < 4; ++mt) {
      v16h a = load_a_f16(Qh + mt * 16 * CH + h * HD, CH); // invariant over ntt
#pragma unroll
      for (int ntt = 0; ntt < 4; ++ntt) {
        v8f acc;
#pragma unroll
        for (int r = 0; r < 8; ++r) acc[r] = 0.0f;
        v16h bw = load_b_t(Kh + ntt * 16 * CH + h * HD, CH);
        acc = wmma16(a, bw, acc);
#pragma unroll
        for (int r = 0; r < 8; ++r) {
          int n = mt * 16 + rbase + r;
          int m = ntt * 16 + col;
          aB[n * TOK + m] = acc[r] + rb_h[n * TOK + m] + mk[n * TOK + m];
        }
      }
    }
  }
  asm volatile("s_wait_dscnt 0" ::: "memory"); // wave-local LDS RAW fence

  // Softmax: per wave on its own [64][64] block, 2 rows per lane
  {
#pragma unroll
    for (int rr = 0; rr < 2; ++rr) {
      const int n = lane * 2 + rr;
      float* rowp = aBuf + wave * TOK * TOK + n * TOK;
      float mx = -3.0e38f;
      for (int m = 0; m < TOK; ++m) mx = fmaxf(mx, rowp[m]);
      float s = 0.0f;
      for (int m = 0; m < TOK; ++m) {
        float e = __expf(rowp[m] - mx);
        rowp[m] = e;
        s += e;
      }
      float inv = 1.0f / s;
      for (int m = 0; m < TOK; ++m) rowp[m] *= inv;
    }
  }
  asm volatile("s_wait_dscnt 0" ::: "memory");

  // Stage D: O_h = P * V_h   (M=64, N=32, K=64) -> Xh reused as O (f16 [64][256])
  {
    const int h = wave;
    const float* aB = aBuf + h * TOK * TOK;
#pragma unroll 1
    for (int mt = 0; mt < 4; ++mt) {
      v16h aF[2];
#pragma unroll
      for (int kk = 0; kk < 2; ++kk)
        aF[kk] = load_a_f32(aB + mt * 16 * TOK + kk * 32, TOK); // hoisted over ntt
#pragma unroll
      for (int ntt = 0; ntt < 2; ++ntt) {
        v8f acc;
#pragma unroll
        for (int r = 0; r < 8; ++r) acc[r] = 0.0f;
#pragma unroll
        for (int kk = 0; kk < 2; ++kk) {
          v16h bw = load_b_t(Vt + (h * HD + ntt * 16) * TOK + kk * 32, TOK);
          acc = wmma16(aF[kk], bw, acc);
        }
        store_d_f16(Xh + mt * 16 * CH + h * HD + ntt * 16, CH, acc, 1.0f);
      }
    }
  }
  __syncthreads();

  // Stage E: out = O * proj_w^T + proj_b  (M=64, N=256, K=256), fp32 to global.
  // mt outer with resident A-fragments; each wave owns 2 N-tiles (rolled loop).
  {
    float* ob = out + (size_t)b * TOK * CH;
#pragma unroll 1
    for (int mt = 0; mt < 4; ++mt) {
      v16h aF[8];
#pragma unroll
      for (int kk = 0; kk < 8; ++kk)
        aF[kk] = load_a_f16(Xh + mt * 16 * CH + kk * 32, CH);
#pragma unroll 1
      for (int ni = 0; ni < 2; ++ni) {
        const int nt = wave * 2 + ni;
        const float bias = projb[nt * 16 + col];
        v8f acc;
#pragma unroll
        for (int r = 0; r < 8; ++r) acc[r] = bias;
#pragma unroll
        for (int kk = 0; kk < 8; ++kk) {
          v16h bw = load_b_t(projw + (size_t)(nt * 16) * CH + kk * 32, CH);
          acc = wmma16(aF[kk], bw, acc);
        }
#pragma unroll
        for (int r = 0; r < 8; ++r) {
          ob[(mt * 16 + rbase + r) * CH + nt * 16 + col] = acc[r];
        }
      }
    }
  }
}

extern "C" void kernel_launch(void* const* d_in, const int* in_sizes, int n_in,
                              void* d_out, int out_size, void* d_ws, size_t ws_size,
                              hipStream_t stream) {
  const float* x          = (const float*)d_in[0];
  const float* mask       = (const float*)d_in[1];
  const float* qkv_w      = (const float*)d_in[2];
  const float* qkv_b      = (const float*)d_in[3];
  const float* proj_w     = (const float*)d_in[4];
  const float* proj_b     = (const float*)d_in[5];
  const float* bias_table = (const float*)d_in[6];
  float* out = (float*)d_out;

  // workspace layout: f16 qkv_w (384KB) | f16 proj_w (128KB) | rel_bias f32 (128KB)
  char* ws = (char*)d_ws;
  _Float16* qkvw_h  = (_Float16*)ws;
  _Float16* projw_h = (_Float16*)(ws + (size_t)3 * CH * CH * sizeof(_Float16));
  float*    relBias = (float*)(ws + (size_t)3 * CH * CH * sizeof(_Float16)
                                  + (size_t)CH * CH * sizeof(_Float16));

  const int prep_total = 3 * CH * CH + CH * CH + NH * TOK * TOK; // 294912
  winattn_prep<<<(prep_total + 255) / 256, 256, 0, stream>>>(
      qkv_w, proj_w, bias_table, qkvw_h, projw_h, relBias);

  // LDS: 4 * (64*256 f16) + 8 * (64*64 f32) = 131072 + 131072 = 262144 bytes
  const size_t smem_bytes = (size_t)4 * TOK * CH * sizeof(_Float16)
                          + (size_t)NH * TOK * TOK * sizeof(float);
  winattn_main<<<4096, 256, smem_bytes, stream>>>(
      x, mask, qkvw_h, qkv_b, projw_h, proj_b, relBias, out);
}
